// Net_58265526337911
// MI455X (gfx1250) — compile-verified, matrix operand order checked by reading, not run on hardware
//
#include <hip/hip_runtime.h>

#define N_NODES 50000
#define N_EDGES 500000
#define FIN     128
#define FOUT    512

typedef __bf16 bf16_t;
typedef __attribute__((ext_vector_type(16))) __bf16 v16bf;
typedef __attribute__((ext_vector_type(8)))  __bf16 v8bf;
typedef __attribute__((ext_vector_type(8)))  float  v8f;

union V16 { v16bf v; v8bf h[2]; };

// ---------------------------------------------------------------------------
// 1) prep: zero accumulator, X -> bf16, W -> bf16 transposed (Wt[n][k]),
//    deg init to 1.0 (self loop contribution).
// ---------------------------------------------------------------------------
__global__ void prep_kernel(const float* __restrict__ x, const float* __restrict__ W,
                            float* __restrict__ out, bf16_t* __restrict__ Xb,
                            bf16_t* __restrict__ Wt, float* __restrict__ deg) {
    size_t tid = (size_t)blockIdx.x * blockDim.x + threadIdx.x;
    if (tid < (size_t)N_NODES * FOUT) out[tid] = 0.0f;
    if (tid < (size_t)N_NODES * FIN)  Xb[tid] = (bf16_t)x[tid];
    if (tid < (size_t)FIN * FOUT) {
        int n = (int)(tid >> 7);      // output channel
        int k = (int)(tid & 127);     // input channel
        Wt[tid] = (bf16_t)W[(size_t)k * FOUT + n];
    }
    if (tid < N_NODES) deg[tid] = 1.0f;
}

// ---------------------------------------------------------------------------
// 2) degree count over target nodes (col)
// ---------------------------------------------------------------------------
__global__ void deg_kernel(const int* __restrict__ col, float* __restrict__ deg) {
    int e = blockIdx.x * blockDim.x + threadIdx.x;
    if (e < N_EDGES) atomicAdd(&deg[col[e]], 1.0f);
}

// ---------------------------------------------------------------------------
// 3) deg_inv_sqrt (deg >= 1 always because of self loops)
// ---------------------------------------------------------------------------
__global__ void dis_kernel(const float* __restrict__ deg, float* __restrict__ dis) {
    int i = blockIdx.x * blockDim.x + threadIdx.x;
    if (i < N_NODES) dis[i] = rsqrtf(deg[i]);
}

// ---------------------------------------------------------------------------
// 4) xw = X @ W via v_wmma_f32_16x16x32_bf16.
//    One wave per 16x16 output tile; K=128 in 4 steps of 32.
//    A layout (ISA 7.12.2, 16-bit A 16x32): lane = row M (lane&15),
//      half-wave selects K offset 0 or 8; VGPR0-3 hold K=kb..kb+7,
//      VGPR4-7 hold K=kb+16..kb+23  -> two contiguous 16B chunks.
//    B is symmetric with N across lanes; Wt is stored [n][k] so the
//    same contiguous-K loads apply.
// ---------------------------------------------------------------------------
__global__ void gemm_wmma_kernel(const bf16_t* __restrict__ Xb,
                                 const bf16_t* __restrict__ Wt,
                                 bf16_t* __restrict__ xwb) {
    const int lane = threadIdx.x & 31;
    const int wave = threadIdx.x >> 5;
    const int t    = blockIdx.x * 8 + wave;   // tile id, exactly 100000 tiles
    const int tn   = t & 31;                  // FOUT/16 = 32
    const int tm   = t >> 5;                  // N/16    = 3125
    const int mn   = lane & 15;               // A: row M ; B: col N
    const int kh   = (lane >> 4) << 3;        // half-wave K offset 0 / 8

    const bf16_t* arow = Xb + (size_t)(tm * 16 + mn) * FIN;
    const bf16_t* bcol = Wt + (size_t)(tn * 16 + mn) * FIN;

    v8f acc = {};
#pragma unroll
    for (int s = 0; s < 4; ++s) {
        const int kb = s * 32 + kh;
        V16 a, b;
        a.h[0] = *(const v8bf*)(arow + kb);
        a.h[1] = *(const v8bf*)(arow + kb + 16);
        b.h[0] = *(const v8bf*)(bcol + kb);
        b.h[1] = *(const v8bf*)(bcol + kb + 16);
        acc = __builtin_amdgcn_wmma_f32_16x16x32_bf16(
            /*neg_a=*/false, a.v, /*neg_b=*/false, b.v,
            /*c_mod=*/(short)0, acc, /*reuse_a=*/false, /*reuse_b=*/false);
    }

    // C/D layout: VGPR r -> M = r (lanes 0-15) or 8+r (lanes 16-31); N = lane&15
    const int rbase = (lane >> 4) << 3;
    bf16_t* dst = xwb + (size_t)(tm * 16) * FOUT + (size_t)(tn * 16 + mn);
#pragma unroll
    for (int r = 0; r < 8; ++r)
        dst[(size_t)(rbase + r) * FOUT] = (bf16_t)acc[r];
}

// ---------------------------------------------------------------------------
// 5) edge scatter: one wave per edge (incl. self loops).
//    Lane covers f = j*32 + lane -> fully coalesced loads + f32 atomics.
// ---------------------------------------------------------------------------
__global__ void scatter_kernel(const int* __restrict__ ei,       // [2*E] row then col
                               const float* __restrict__ dis,
                               const bf16_t* __restrict__ xwb,
                               float* __restrict__ agg) {
    const int lane = threadIdx.x & 31;
    const int wave = threadIdx.x >> 5;
    const int e    = blockIdx.x * 8 + wave;
    const int EE   = N_EDGES + N_NODES;
    if (e >= EE) return;

    int r, c;
    if (e < N_EDGES) { r = ei[e]; c = ei[N_EDGES + e]; }
    else             { r = c = e - N_EDGES; }            // self loop

    const float w = dis[r] * dis[c];
    const bf16_t* src = xwb + (size_t)r * FOUT;
    float*        dst = agg + (size_t)c * FOUT;
#pragma unroll
    for (int j = 0; j < 16; ++j) {
        const int f = j * 32 + lane;
        atomicAdd(dst + f, w * (float)src[f]);
    }
}

// ---------------------------------------------------------------------------
// 6) out = relu(agg + b)
// ---------------------------------------------------------------------------
__global__ void fin_kernel(float* __restrict__ out, const float* __restrict__ b) {
    size_t t = (size_t)blockIdx.x * blockDim.x + threadIdx.x;
    if (t < (size_t)N_NODES * FOUT)
        out[t] = fmaxf(out[t] + b[t & (FOUT - 1)], 0.0f);
}

// ---------------------------------------------------------------------------
extern "C" void kernel_launch(void* const* d_in, const int* in_sizes, int n_in,
                              void* d_out, int out_size, void* d_ws, size_t ws_size,
                              hipStream_t stream) {
    const float* x  = (const float*)d_in[0];   // [N, 128]
    const int*   ei = (const int*)d_in[1];     // [2, E] (jax default int32)
    const float* W  = (const float*)d_in[2];   // [128, 512]
    const float* b  = (const float*)d_in[3];   // [512]
    float* out = (float*)d_out;                // [N, 512]

    // workspace layout (all offsets 64B aligned)
    char* base = (char*)d_ws;
    bf16_t* Xb  = (bf16_t*)(base);                               // 12,800,000 B
    bf16_t* Wt  = (bf16_t*)(base + 12800000);                    //    131,072 B
    bf16_t* xwb = (bf16_t*)(base + 12800000 + 131072);           // 51,200,000 B
    float*  deg = (float*) (base + 12800000 + 131072 + 51200000);//    200,000 B
    float*  dis = (float*) (base + 12800000 + 131072 + 51200000 + 200000);

    const int THREADS = 256;

    // 1) init / convert
    prep_kernel<<<(N_NODES * FOUT + THREADS - 1) / THREADS, THREADS, 0, stream>>>(
        x, W, out, Xb, Wt, deg);
    // 2) degree
    deg_kernel<<<(N_EDGES + THREADS - 1) / THREADS, THREADS, 0, stream>>>(
        ei + N_EDGES, deg);
    // 3) deg^-1/2
    dis_kernel<<<(N_NODES + THREADS - 1) / THREADS, THREADS, 0, stream>>>(deg, dis);
    // 4) GEMM via WMMA: 100000 tiles, 8 waves/block -> 12500 blocks
    gemm_wmma_kernel<<<12500, THREADS, 0, stream>>>(Xb, Wt, xwb);
    // 5) edge gather/scale/scatter-add: 550000 edges, 8 per block
    scatter_kernel<<<(N_EDGES + N_NODES + 7) / 8, THREADS, 0, stream>>>(
        ei, dis, xwb, out);
    // 6) bias + relu
    fin_kernel<<<(N_NODES * FOUT + THREADS - 1) / THREADS, THREADS, 0, stream>>>(out, b);
}
